// Model_82566451299120
// MI455X (gfx1250) — compile-verified, matrix operand order checked by reading, not run on hardware
//
#include <hip/hip_runtime.h>
#include <hip/hip_bf16.h>
#include <cstdint>
#include <cstddef>

// ---------------------------------------------------------------------------
// Problem constants (from the reference)
// ---------------------------------------------------------------------------
constexpr int Bq   = 256;        // batch
constexpr int Lq   = 192;        // encoder sequence length
constexpr int Eq   = 32;         // input features
constexpr int Dq   = 1024;       // GRU hidden
constexpr int PRED = 96;         // decoder steps
constexpr int OUTD = 32;         // decoder output features
constexpr int KMA  = 25;         // moving-average kernel
constexpr int PAD  = (KMA - 1) / 2;

typedef __bf16 bf16_t;
typedef __attribute__((ext_vector_type(16))) __bf16 v16bf;
typedef __attribute__((ext_vector_type(8)))  float  v8f;

__device__ __forceinline__ bf16_t f2bf(float f) {
  uint32_t u = __builtin_bit_cast(uint32_t, f);
  uint32_t r = u + 0x7FFFu + ((u >> 16) & 1u);   // round-to-nearest-even
  uint16_t h = (uint16_t)(r >> 16);
  return __builtin_bit_cast(bf16_t, h);
}

__device__ __forceinline__ float sigmoidf_(float x) {
  return 1.f / (1.f + __expf(-x));
}

// ---------------------------------------------------------------------------
// fp32 -> bf16 weight conversion
// ---------------------------------------------------------------------------
__global__ __launch_bounds__(256) void cvt_bf16_kernel(
    const float* __restrict__ s, bf16_t* __restrict__ d, int n) {
  int i = blockIdx.x * 256 + threadIdx.x;
  if (i < n) d[i] = f2bf(s[i]);
}

// ---------------------------------------------------------------------------
// series_decomp: per (b,e) thread, clamped 25-wide moving average
// (replicate padding == index clamp). Emits bf16 encoder inputs laid out
// [t][B][E] and the last timestep as the decoder's first bf16 inputs.
// ---------------------------------------------------------------------------
__global__ __launch_bounds__(256) void decomp_kernel(
    const float* __restrict__ x_enc,
    bf16_t* __restrict__ encS_in, bf16_t* __restrict__ encT_in,
    bf16_t* __restrict__ xs0, bf16_t* __restrict__ xt0) {
  const int i = blockIdx.x * 256 + threadIdx.x;   // over B*E
  if (i >= Bq * Eq) return;
  const int b = i / Eq, e = i % Eq;
  const float* xb = x_enc + (size_t)b * Lq * Eq + e;
  for (int t = 0; t < Lq; ++t) {
    float s = 0.f;
    #pragma unroll 5
    for (int j = t - PAD; j <= t + PAD; ++j) {
      int jj = j < 0 ? 0 : (j > Lq - 1 ? Lq - 1 : j);
      s += xb[(size_t)jj * Eq];
    }
    const float mean   = s * (1.0f / (float)KMA);
    const float season = xb[(size_t)t * Eq] - mean;
    if (t < Lq - 1) {
      encS_in[(size_t)t * Bq * Eq + b * Eq + e] = f2bf(season);
      encT_in[(size_t)t * Bq * Eq + b * Eq + e] = f2bf(mean);
    } else {
      xs0[b * Eq + e] = f2bf(season);
      xt0[b * Eq + e] = f2bf(mean);
    }
  }
}

// ---------------------------------------------------------------------------
// Fully fused GRU layer step (WMMA bf16 16x16x32, f32 accumulate):
//
//   r = sigmoid(X·Wih_rᵀ + H·Whh_rᵀ + bih_r + bhh_r)   <- one shared acc
//   z = sigmoid(X·Wih_zᵀ + H·Whh_zᵀ + bih_z + bhh_z)   <- one shared acc
//   n = tanh( (X·Wih_nᵀ+bih_n) + r*(H·Whh_nᵀ+bhh_n) )  <- two accs
//   h' = (1-z)*n + z*h
//
// Wave tile: 16 batch-rows x 32 hidden-cols (2 col tiles x 4 gate-role accs
// = 8 v8f accumulators). Block: 8 waves -> 64x64. Grid (1024/64, 256/64)=64.
// A-fragment (this code's contiguous k order, B uses the same): lane = 16*(k>=16) + m, 16 contiguous bf16 of row m.
// B[k][n] = W[n][k] -> lane ln loads 16 contiguous bf16 from W row n0+ln.
// C layout: row = m0 + 8*(lane>>4) + v, col = base + (lane&15).
//
// Hin (bf16 state) is read whole-row by every block -> must be a different
// buffer from Hout (ping-pong). f32 carry h is elementwise-owned -> in-place.
// ---------------------------------------------------------------------------
__global__ __launch_bounds__(256) void gru_step_fused_kernel(
    const bf16_t* __restrict__ X, int Kx,
    const bf16_t* __restrict__ Hin,
    const bf16_t* __restrict__ Wih, const bf16_t* __restrict__ Whh,
    const float* __restrict__ bih, const float* __restrict__ bhh,
    float* __restrict__ h, bf16_t* __restrict__ Hout) {
  const int lane = threadIdx.x & 31;
  const int w    = threadIdx.x >> 5;
  const int m0   = blockIdx.y * 64 + (w & 3) * 16;
  const int d0   = blockIdx.x * 64 + (w >> 2) * 32;
  const int ln   = lane & 15;
  const int koff = (lane >> 4) * 16;
  const int aRow = m0 + ln;
  const int rb   = m0 + (lane >> 4) * 8;
  const v8f zero = {0.f, 0.f, 0.f, 0.f, 0.f, 0.f, 0.f, 0.f};

  v8f accR[2], accZ[2], accIN[2], accHN[2];
  #pragma unroll
  for (int j = 0; j < 2; ++j) {
    accR[j] = zero; accZ[j] = zero; accIN[j] = zero; accHN[j] = zero;
  }

  // ---- pass 1: X · Wihᵀ (r, z, gi_n) --------------------------------------
  for (int k = 0; k < Kx; k += 32) {
    v16bf a = *reinterpret_cast<const v16bf*>(X + (size_t)aRow * Kx + k + koff);
    #pragma unroll
    for (int j = 0; j < 2; ++j) {
      const int col = d0 + j * 16 + ln;
      v16bf br = *reinterpret_cast<const v16bf*>(Wih + (size_t)(col         ) * Kx + k + koff);
      v16bf bz = *reinterpret_cast<const v16bf*>(Wih + (size_t)(col +     Dq) * Kx + k + koff);
      v16bf bn = *reinterpret_cast<const v16bf*>(Wih + (size_t)(col + 2 * Dq) * Kx + k + koff);
      accR[j]  = __builtin_amdgcn_wmma_f32_16x16x32_bf16(false, a, false, br, (short)0, accR[j],  false, false);
      accZ[j]  = __builtin_amdgcn_wmma_f32_16x16x32_bf16(false, a, false, bz, (short)0, accZ[j],  false, false);
      accIN[j] = __builtin_amdgcn_wmma_f32_16x16x32_bf16(false, a, false, bn, (short)0, accIN[j], false, false);
    }
  }

  // ---- pass 2: H · Whhᵀ (r, z accumulate on top; gh_n separate) -----------
  for (int k = 0; k < Dq; k += 32) {
    v16bf a = *reinterpret_cast<const v16bf*>(Hin + (size_t)aRow * Dq + k + koff);
    #pragma unroll
    for (int j = 0; j < 2; ++j) {
      const int col = d0 + j * 16 + ln;
      v16bf br = *reinterpret_cast<const v16bf*>(Whh + (size_t)(col         ) * Dq + k + koff);
      v16bf bz = *reinterpret_cast<const v16bf*>(Whh + (size_t)(col +     Dq) * Dq + k + koff);
      v16bf bn = *reinterpret_cast<const v16bf*>(Whh + (size_t)(col + 2 * Dq) * Dq + k + koff);
      accR[j]  = __builtin_amdgcn_wmma_f32_16x16x32_bf16(false, a, false, br, (short)0, accR[j],  false, false);
      accZ[j]  = __builtin_amdgcn_wmma_f32_16x16x32_bf16(false, a, false, bz, (short)0, accZ[j],  false, false);
      accHN[j] = __builtin_amdgcn_wmma_f32_16x16x32_bf16(false, a, false, bn, (short)0, accHN[j], false, false);
    }
  }

  // ---- epilogue: gates + state update, all in registers -------------------
  #pragma unroll
  for (int j = 0; j < 2; ++j) {
    const int col = d0 + j * 16 + ln;
    const float br  = bih[col]          + bhh[col];
    const float bz  = bih[col + Dq]     + bhh[col + Dq];
    const float bin = bih[col + 2 * Dq];
    const float bhn = bhh[col + 2 * Dq];
    #pragma unroll
    for (int v = 0; v < 8; ++v) {
      const size_t idx = (size_t)(rb + v) * Dq + col;
      const float rg = sigmoidf_(accR[j][v] + br);
      const float zg = sigmoidf_(accZ[j][v] + bz);
      const float ng = tanhf(accIN[j][v] + bin + rg * (accHN[j][v] + bhn));
      const float hv = (1.f - zg) * ng + zg * h[idx];
      h[idx]    = hv;
      Hout[idx] = f2bf(hv);
    }
  }
}

// ---------------------------------------------------------------------------
// Decoder tail, fused: os = Hs·Wsᵀ+bs ; ot = Ht·Wtᵀ+bt ; out[b][t][:] = os+ot
// and re-emit os/ot as bf16 next-step decoder inputs. 32 tiles of 16x16 per
// branch; each wave computes one tile of BOTH branches. 4 blocks x 8 waves.
// ---------------------------------------------------------------------------
__global__ __launch_bounds__(256) void proj_combine_kernel(
    const bf16_t* __restrict__ Hs, const bf16_t* __restrict__ Ht,
    const bf16_t* __restrict__ Ws, const bf16_t* __restrict__ Wt,
    const float* __restrict__ bs, const float* __restrict__ bt,
    float* __restrict__ out, int t,
    bf16_t* __restrict__ xsb, bf16_t* __restrict__ xtb) {
  const int lane = threadIdx.x & 31;
  const int wv   = threadIdx.x >> 5;
  const int tile = blockIdx.x * 8 + wv;           // 0..31
  const int m0   = (tile & 15) * 16;
  const int n0   = (tile >> 4) * 16;
  const int ln   = lane & 15;
  const int koff = (lane >> 4) * 16;
  const v8f zero = {0.f, 0.f, 0.f, 0.f, 0.f, 0.f, 0.f, 0.f};
  v8f accS = zero, accT = zero;
  for (int k = 0; k < Dq; k += 32) {
    v16bf as  = *reinterpret_cast<const v16bf*>(Hs + (size_t)(m0 + ln) * Dq + k + koff);
    v16bf bsw = *reinterpret_cast<const v16bf*>(Ws + (size_t)(n0 + ln) * Dq + k + koff);
    accS = __builtin_amdgcn_wmma_f32_16x16x32_bf16(false, as, false, bsw, (short)0, accS, false, false);
    v16bf at  = *reinterpret_cast<const v16bf*>(Ht + (size_t)(m0 + ln) * Dq + k + koff);
    v16bf btw = *reinterpret_cast<const v16bf*>(Wt + (size_t)(n0 + ln) * Dq + k + koff);
    accT = __builtin_amdgcn_wmma_f32_16x16x32_bf16(false, at, false, btw, (short)0, accT, false, false);
  }
  const int rb  = m0 + (lane >> 4) * 8;
  const int col = n0 + ln;
  const float bsv = bs[col], btv = bt[col];
  #pragma unroll
  for (int v = 0; v < 8; ++v) {
    const int row = rb + v;
    const float s  = accS[v] + bsv;
    const float tr = accT[v] + btv;
    out[(size_t)row * (PRED * OUTD) + t * OUTD + col] = s + tr;
    xsb[row * OUTD + col] = f2bf(s);
    xtb[row * OUTD + col] = f2bf(tr);
  }
}

// ---------------------------------------------------------------------------
// Host orchestration
// ---------------------------------------------------------------------------
extern "C" void kernel_launch(void* const* d_in, const int* in_sizes, int n_in,
                              void* d_out, int out_size, void* d_ws, size_t ws_size,
                              hipStream_t stream) {
  (void)in_sizes; (void)n_in; (void)out_size; (void)ws_size;
  auto F = [&](int i) { return (const float*)d_in[i]; };

  // Input order (setup_inputs insertion order, pytrees flattened in order):
  // 0:x_enc 1:x_dec(unused)
  // per GRU (enc_s, dec_s, enc_t, dec_t), per layer: Wih, Whh, bih, bhh
  struct GruP { const float *Wih0, *Whh0, *bih0, *bhh0, *Wih1, *Whh1, *bih1, *bhh1; };
  GruP P[4];
  for (int g = 0; g < 4; ++g) {
    int o = 2 + g * 8;
    P[g] = { F(o+0), F(o+1), F(o+2), F(o+3), F(o+4), F(o+5), F(o+6), F(o+7) };
  }
  const float *Ws = F(34), *bs = F(35), *Wt = F(36), *bt = F(37);
  const float* x_enc = F(0);
  float* out = (float*)d_out;

  // ---- bump allocator over d_ws -------------------------------------------
  size_t off = 0;
  auto alloc = [&](size_t bytes) -> void* {
    void* p = (char*)d_ws + off;
    off += (bytes + 255) & ~(size_t)255;
    return p;
  };

  bf16_t* encS_in = (bf16_t*)alloc((size_t)(Lq - 1) * Bq * Eq * sizeof(bf16_t));
  bf16_t* encT_in = (bf16_t*)alloc((size_t)(Lq - 1) * Bq * Eq * sizeof(bf16_t));

  // bf16 weights: [g][Wih0, Whh0, Wih1, Whh1]
  bf16_t* Wb[4][4];
  const size_t wsz[4] = { (size_t)3 * Dq * Eq, (size_t)3 * Dq * Dq,
                          (size_t)3 * Dq * Dq, (size_t)3 * Dq * Dq };
  for (int g = 0; g < 4; ++g)
    for (int k = 0; k < 4; ++k)
      Wb[g][k] = (bf16_t*)alloc(wsz[k] * sizeof(bf16_t));
  bf16_t* WsB = (bf16_t*)alloc((size_t)OUTD * Dq * sizeof(bf16_t));
  bf16_t* WtB = (bf16_t*)alloc((size_t)OUTD * Dq * sizeof(bf16_t));

  // Hidden states: slots 0,1 = seasonal L0,L1; slots 2,3 = trend L0,L1.
  // f32 carry (in-place safe) + ping-pong bf16 copies (cross-block read).
  const size_t hN = (size_t)Bq * Dq;
  float*  h[4];
  bf16_t* hbf[4][2];
  int     par[4] = {0, 0, 0, 0};
  char* hBlk   = (char*)alloc(4 * hN * sizeof(float));
  char* hbfBlk = (char*)alloc(8 * hN * sizeof(bf16_t));
  for (int i = 0; i < 4; ++i) {
    h[i]      = (float*)(hBlk + (size_t)i * hN * sizeof(float));
    hbf[i][0] = (bf16_t*)(hbfBlk + (size_t)(2 * i)     * hN * sizeof(bf16_t));
    hbf[i][1] = (bf16_t*)(hbfBlk + (size_t)(2 * i + 1) * hN * sizeof(bf16_t));
  }

  bf16_t* xsb = (bf16_t*)alloc((size_t)Bq * OUTD * sizeof(bf16_t));
  bf16_t* xtb = (bf16_t*)alloc((size_t)Bq * OUTD * sizeof(bf16_t));

  // ---- one-time prep -------------------------------------------------------
  hipMemsetAsync(hBlk,   0, 4 * hN * sizeof(float),  stream);
  hipMemsetAsync(hbfBlk, 0, 8 * hN * sizeof(bf16_t), stream);

  auto cvt = [&](const float* s, bf16_t* d, size_t n) {
    cvt_bf16_kernel<<<dim3((unsigned)((n + 255) / 256)), 256, 0, stream>>>(s, d, (int)n);
  };
  for (int g = 0; g < 4; ++g) {
    cvt(P[g].Wih0, Wb[g][0], wsz[0]);
    cvt(P[g].Whh0, Wb[g][1], wsz[1]);
    cvt(P[g].Wih1, Wb[g][2], wsz[2]);
    cvt(P[g].Whh1, Wb[g][3], wsz[3]);
  }
  cvt(Ws, WsB, (size_t)OUTD * Dq);
  cvt(Wt, WtB, (size_t)OUTD * Dq);

  decomp_kernel<<<dim3((Bq * Eq) / 256), 256, 0, stream>>>(
      x_enc, encS_in, encT_in, xsb, xtb);

  // ---- fused GRU layer step ------------------------------------------------
  const dim3 gGrid(Dq / 64, Bq / 64);   // (16, 4)
  auto layer_step = [&](const bf16_t* x, int Kx, int g, int layer,
                        int slot) -> const bf16_t* {
    const bf16_t* Wihb = Wb[g][layer == 0 ? 0 : 2];
    const bf16_t* Whhb = Wb[g][layer == 0 ? 1 : 3];
    const float*  bih  = layer == 0 ? P[g].bih0 : P[g].bih1;
    const float*  bhh  = layer == 0 ? P[g].bhh0 : P[g].bhh1;
    bf16_t* hin  = hbf[slot][par[slot]];
    bf16_t* hout = hbf[slot][par[slot] ^ 1];
    gru_step_fused_kernel<<<gGrid, 256, 0, stream>>>(
        x, Kx, hin, Wihb, Whhb, bih, bhh, h[slot], hout);
    par[slot] ^= 1;
    return hout;
  };

  // ---- encoders (191 sequential steps, seasonal + trend branches) ----------
  for (int t = 0; t < Lq - 1; ++t) {
    const bf16_t* xs = encS_in + (size_t)t * Bq * Eq;
    const bf16_t* xt = encT_in + (size_t)t * Bq * Eq;
    const bf16_t* s0 = layer_step(xs, Eq, /*enc_s*/0, 0, 0);
    layer_step(s0, Dq, /*enc_s*/0, 1, 1);
    const bf16_t* t0 = layer_step(xt, Eq, /*enc_t*/2, 0, 2);
    layer_step(t0, Dq, /*enc_t*/2, 1, 3);
  }

  // ---- decoders (96 sequential steps; hidden state continues from enc) -----
  for (int t = 0; t < PRED; ++t) {
    const bf16_t* s0    = layer_step(xsb, Eq, /*dec_s*/1, 0, 0);
    const bf16_t* sTop  = layer_step(s0,  Dq, /*dec_s*/1, 1, 1);
    const bf16_t* t0    = layer_step(xtb, Eq, /*dec_t*/3, 0, 2);
    const bf16_t* tTop  = layer_step(t0,  Dq, /*dec_t*/3, 1, 3);
    proj_combine_kernel<<<dim3(4), 256, 0, stream>>>(
        sTop, tTop, WsB, WtB, bs, bt, out, t, xsb, xtb);
  }
}
